// CrossAttention_61357902791349
// MI455X (gfx1250) — matrix operand order repaired and checked by reading, not compiled, not run on hardware
//
#include <hip/hip_runtime.h>

// ---- problem constants (match reference) ----
#define NB    2
#define NQ    5376
#define MK    512
#define BDIM  1536
#define HEADS 12
#define HDIM  128
#define RQ    (NB*NQ)      // 10752 rows for Q / out GEMMs
#define RKV   (NB*MK)      // 1024 rows for KV GEMM

typedef __bf16 bf16;
typedef __attribute__((ext_vector_type(16))) __bf16 v16bf;
typedef __attribute__((ext_vector_type(8)))  __bf16 v8bf;
typedef __attribute__((ext_vector_type(8)))  float  v8f;
typedef __attribute__((ext_vector_type(4)))  int    v4i;

#if defined(__has_builtin)
#if __has_builtin(__builtin_amdgcn_global_load_async_to_lds_b128) && \
    __has_builtin(__builtin_amdgcn_s_wait_asynccnt)
#define USE_ASYNC_LDS 1
#endif
#endif
#ifndef USE_ASYNC_LDS
#define USE_ASYNC_LDS 0
#endif

#if USE_ASYNC_LDS
typedef __attribute__((address_space(1))) v4i g_v4i;   // global
typedef __attribute__((address_space(3))) v4i s_v4i;   // LDS
#endif

__device__ __forceinline__ v16bf ld16(const bf16* p0, const bf16* p1) {
    v8bf a = *(const v8bf*)p0;
    v8bf b = *(const v8bf*)p1;
    v16bf r;
#pragma unroll
    for (int i = 0; i < 8; ++i) { r[i] = a[i]; r[i + 8] = b[i]; }
    return r;
}

__device__ __forceinline__ v8f wmma_bf16(v16bf a, v16bf b, v8f c) {
    return __builtin_amdgcn_wmma_f32_16x16x32_bf16(false, a, false, b, (short)0, c, false, false);
}

// copy 16 bf16 (two b128) global -> LDS, async if available
__device__ __forceinline__ void cp16(const bf16* src, bf16* dst) {
#if USE_ASYNC_LDS
    __builtin_amdgcn_global_load_async_to_lds_b128((g_v4i*)src,       (s_v4i*)dst,       0, 0);
    __builtin_amdgcn_global_load_async_to_lds_b128((g_v4i*)(src + 8), (s_v4i*)(dst + 8), 0, 0);
#else
    uint4 a0 = *(const uint4*)src;
    uint4 a1 = *(const uint4*)(src + 8);
    *(uint4*)dst = a0;
    *(uint4*)(dst + 8) = a1;
#endif
}

__device__ __forceinline__ void wait_stage() {
#if USE_ASYNC_LDS
    __builtin_amdgcn_s_wait_asynccnt(0);
#endif
}

// ---------------- conversion kernels ----------------
__global__ void cast_bf16_kernel(const float* __restrict__ s, bf16* __restrict__ d, int n) {
    int i = (blockIdx.x * blockDim.x + threadIdx.x) * 4;
    if (i + 3 < n) {
        float4 v = *(const float4*)(s + i);
        d[i + 0] = (bf16)v.x; d[i + 1] = (bf16)v.y;
        d[i + 2] = (bf16)v.z; d[i + 3] = (bf16)v.w;
    } else {
        for (; i < n; ++i) d[i] = (bf16)s[i];
    }
}

// wt[c*K + k] = w[k*C + c]
__global__ void transpose_bf16_kernel(const float* __restrict__ w, bf16* __restrict__ wt,
                                      int K, int C) {
    int idx = blockIdx.x * blockDim.x + threadIdx.x;
    if (idx < K * C) {
        int c = idx / K, k = idx - c * K;
        wt[idx] = (bf16)w[(size_t)k * C + c];
    }
}

// ---------------- WMMA GEMM, 128x128 tile, K-step 32, double-buffered LDS --------
// MODE 0: Q projection: bias + per-head RMSNorm + q_norm_scale -> bf16 [B,H,N,HD]
// MODE 1: KV projection: bias -> K bf16 [B,H,M,HD], V bf16 transposed [B,H,HD,M]
// MODE 2: output projection: bias -> f32 [rows, totCols]
template <int MODE>
__global__ __launch_bounds__(256) void gemm_wmma(
    const bf16* __restrict__ A,    // [rows, K] row-major
    const bf16* __restrict__ Wt,   // [totCols, K] (pre-transposed)
    const float* __restrict__ bias,
    int K, int totCols, int rowsPerBatch,
    float* __restrict__ outF, bf16* __restrict__ outQ,
    const float* __restrict__ qns,
    bf16* __restrict__ outK, bf16* __restrict__ outVt)
{
    __shared__ bf16 As[2][128 * 32];
    __shared__ bf16 Bs[2][128 * 32];
    __shared__ float red[2][128];

    const int row0 = blockIdx.x * 128;
    const int col0 = blockIdx.y * 128;
    const int tid  = threadIdx.x;
    const int lane = tid & 31;
    const int wave = tid >> 5;
    const int mrow0 = (wave & 3) * 32;   // 2 M-tiles of 16 rows each
    const int nc0   = (wave >> 2) * 64;  // 4 N-tiles of 16 cols each
    const int l15 = lane & 15;
    const bool hi = lane >= 16;

    // staging index: 256 threads x 16 bf16 = one 128x32 tile
    const int sIdx = tid * 16;
    const int sR  = sIdx >> 5;      // 0..127
    const int sKK = sIdx & 31;      // 0 or 16

    v8f c[2][4];
#pragma unroll
    for (int m = 0; m < 2; ++m)
#pragma unroll
        for (int t = 0; t < 4; ++t) c[m][t] = 0.0f;

    // prologue: fill buffer 0
    cp16(&A [(size_t)(row0 + sR) * K + sKK], &As[0][sR * 32 + sKK]);
    cp16(&Wt[(size_t)(col0 + sR) * K + sKK], &Bs[0][sR * 32 + sKK]);
    wait_stage();
    __syncthreads();

    const int abase = hi ? 8 : 0;
    const int bbase = hi ? 8 : 0;

    for (int k0 = 0; k0 < K; k0 += 32) {
        const int cur = (k0 >> 5) & 1;
        if (k0 + 32 < K) {   // prefetch next K-slice into other buffer
            cp16(&A [(size_t)(row0 + sR) * K + k0 + 32 + sKK], &As[cur ^ 1][sR * 32 + sKK]);
            cp16(&Wt[(size_t)(col0 + sR) * K + k0 + 32 + sKK], &Bs[cur ^ 1][sR * 32 + sKK]);
        }
        const bf16* as = As[cur];
        const bf16* bs = Bs[cur];
        const int ar0 = (mrow0 + l15) * 32;
        const int ar1 = (mrow0 + 16 + l15) * 32;
        v16bf af0 = ld16(&as[ar0 + abase], &as[ar0 + 16 + abase]);
        v16bf af1 = ld16(&as[ar1 + abase], &as[ar1 + 16 + abase]);
#pragma unroll
        for (int t = 0; t < 4; ++t) {
            const int bc = (nc0 + t * 16 + l15) * 32;
            v16bf bfr = ld16(&bs[bc + bbase], &bs[bc + bbase + 16]);
            c[0][t] = wmma_bf16(af0, bfr, c[0][t]);
            c[1][t] = wmma_bf16(af1, bfr, c[1][t]);
        }
        wait_stage();
        __syncthreads();
    }

    // ---- epilogue ----
    // C layout: lane<16 -> col=lane, rows v; lane>=16 -> col=lane-16, rows v+8
#pragma unroll
    for (int t = 0; t < 4; ++t) {
        float bv = bias[col0 + nc0 + t * 16 + l15];
#pragma unroll
        for (int m = 0; m < 2; ++m)
#pragma unroll
            for (int v = 0; v < 8; ++v) c[m][t][v] += bv;
    }

    if (MODE == 2) {
#pragma unroll
        for (int m = 0; m < 2; ++m)
#pragma unroll
            for (int t = 0; t < 4; ++t) {
                int col = col0 + nc0 + t * 16 + l15;
#pragma unroll
                for (int v = 0; v < 8; ++v) {
                    int rl = mrow0 + m * 16 + v + (hi ? 8 : 0);
                    outF[(size_t)(row0 + rl) * totCols + col] = c[m][t][v];
                }
            }
    } else if (MODE == 0) {
        // per-row sum of squares over this wave's 64 cols, per M-tile
        float ss[2][8];
#pragma unroll
        for (int m = 0; m < 2; ++m)
#pragma unroll
            for (int v = 0; v < 8; ++v) {
                float s = 0.f;
#pragma unroll
                for (int t = 0; t < 4; ++t) s += c[m][t][v] * c[m][t][v];
                ss[m][v] = s;
            }
#pragma unroll
        for (int off = 1; off < 16; off <<= 1)
#pragma unroll
            for (int m = 0; m < 2; ++m)
#pragma unroll
                for (int v = 0; v < 8; ++v) ss[m][v] += __shfl_xor(ss[m][v], off, 32);
        if (l15 == 0) {
#pragma unroll
            for (int m = 0; m < 2; ++m)
#pragma unroll
                for (int v = 0; v < 8; ++v)
                    red[wave >> 2][mrow0 + m * 16 + v + (hi ? 8 : 0)] = ss[m][v];
        }
        __syncthreads();
        const int h = col0 >> 7;   // block spans exactly one head
#pragma unroll
        for (int m = 0; m < 2; ++m)
#pragma unroll
            for (int v = 0; v < 8; ++v) {
                int rl = mrow0 + m * 16 + v + (hi ? 8 : 0);
                float tot = red[0][rl] + red[1][rl];
                float inv = rsqrtf(tot * (1.0f / HDIM) + 1e-6f);
                int grow = row0 + rl;
                int b = grow / rowsPerBatch, n = grow - b * rowsPerBatch;
                size_t base = (((size_t)b * HEADS + h) * rowsPerBatch + n) * HDIM;
#pragma unroll
                for (int t = 0; t < 4; ++t) {
                    int hd = nc0 + t * 16 + l15;
                    outQ[base + hd] = (bf16)(c[m][t][v] * inv * qns[hd]);
                }
            }
    } else { // MODE 1: KV
        const int sel = col0 / BDIM;
        const int rem = col0 - sel * BDIM;
        const int h = rem >> 7;
#pragma unroll
        for (int t = 0; t < 4; ++t) {
            int hd = (nc0 + t * 16 + l15) & 127;
#pragma unroll
            for (int m = 0; m < 2; ++m)
#pragma unroll
                for (int v = 0; v < 8; ++v) {
                    int rl = mrow0 + m * 16 + v + (hi ? 8 : 0);
                    int grow = row0 + rl;
                    int b = grow / rowsPerBatch, mm = grow - b * rowsPerBatch;
                    bf16 val = (bf16)c[m][t][v];
                    if (sel == 0)
                        outK[(((size_t)b * HEADS + h) * rowsPerBatch + mm) * HDIM + hd] = val;
                    else
                        outVt[(((size_t)b * HEADS + h) * HDIM + hd) * rowsPerBatch + mm] = val;
                }
        }
    }
}

// ---------------- flash attention: one wave per 16 Q rows per (b,h) ----------------
// S^T = K @ Q^T puts q-rows in lanes -> online softmax needs only shfl_xor(.,16),
// and exp() results are already in P's A-fragment layout for P @ V.
__global__ __launch_bounds__(32) void attn_kernel(
    const bf16* __restrict__ q,   // [B,H,N,HD]
    const bf16* __restrict__ k,   // [B,H,M,HD]
    const bf16* __restrict__ vt,  // [B,H,HD,M]
    bf16* __restrict__ aout)      // [B*N, H*HD]
{
    const int lane = threadIdx.x;
    const int l15 = lane & 15;
    const bool hi = lane >= 16;
    const int n0 = blockIdx.x * 16;
    const int h  = blockIdx.y;
    const int b  = blockIdx.z;
    const size_t bh = (size_t)b * HEADS + h;

    const bf16* qbase = q  + (bh * NQ + n0) * HDIM;
    const bf16* kbase = k  + bh * (size_t)MK * HDIM;
    const bf16* vbase = vt + bh * (size_t)HDIM * MK;

    // Q^T B-fragments: 4 chunks of K(d)=32; comps = d 8h..8h+7, 16+8h..16+8h+7
    v16bf qb[4];
    {
        const bf16* qp = qbase + l15 * HDIM;
#pragma unroll
        for (int dc = 0; dc < 4; ++dc) {
            int dbase = dc * 32 + (hi ? 8 : 0);
            qb[dc] = ld16(qp + dbase, qp + dbase + 16);
        }
    }

    float m_run = -1e30f, l_run = 0.0f;
    v8f o[8];
#pragma unroll
    for (int t = 0; t < 8; ++t) o[t] = 0.0f;
    const float scale = 0.08838834764831845f;   // 1/sqrt(128)

    for (int j0 = 0; j0 < MK; j0 += 32) {
        v8f st0 = 0.0f, st1 = 0.0f;
        const bf16* kp0 = kbase + (size_t)(j0 + l15) * HDIM;
        const bf16* kp1 = kbase + (size_t)(j0 + 16 + l15) * HDIM;
#pragma unroll
        for (int dc = 0; dc < 4; ++dc) {
            int off = dc * 32 + (hi ? 8 : 0);
            v16bf ka0 = ld16(kp0 + off, kp0 + off + 16);
            st0 = wmma_bf16(ka0, qb[dc], st0);
            v16bf ka1 = ld16(kp1 + off, kp1 + off + 16);
            st1 = wmma_bf16(ka1, qb[dc], st1);
        }
        // online softmax (lane = q-row, 8+8 keys per lane, partner at lane^16)
        float mc = -1e30f;
#pragma unroll
        for (int i = 0; i < 8; ++i) {
            st0[i] *= scale; st1[i] *= scale;
            mc = fmaxf(mc, fmaxf(st0[i], st1[i]));
        }
        mc = fmaxf(mc, __shfl_xor(mc, 16, 32));
        float m_new = fmaxf(m_run, mc);
        float corr = __expf(m_run - m_new);
        float p0[8], p1[8], ps = 0.f;
#pragma unroll
        for (int i = 0; i < 8; ++i) {
            p0[i] = __expf(st0[i] - m_new);
            p1[i] = __expf(st1[i] - m_new);
            ps += p0[i] + p1[i];
        }
        ps += __shfl_xor(ps, 16, 32);
        l_run = l_run * corr + ps;
        m_run = m_new;
        // rescale O accumulator (rows live in VGPR index; fetch per-row corr)
#pragma unroll
        for (int v = 0; v < 8; ++v) {
            float cv = __shfl(corr, v + (hi ? 8 : 0), 32);
#pragma unroll
            for (int t = 0; t < 8; ++t) o[t][v] *= cv;
        }
        // P A-fragment: exactly the values this lane already holds
        v16bf pa;
#pragma unroll
        for (int i = 0; i < 8; ++i) { pa[i] = (bf16)p0[i]; pa[i + 8] = (bf16)p1[i]; }
        // O += P @ V   (V transposed: contiguous keys per d-column)
#pragma unroll
        for (int t = 0; t < 8; ++t) {
            const bf16* vp = vbase + (size_t)(t * 16 + l15) * MK + j0 + (hi ? 8 : 0);
            v16bf vb = ld16(vp, vp + 16);
            o[t] = wmma_bf16(pa, vb, o[t]);
        }
    }

    // normalize + store bf16 into [B*N, 1536]
#pragma unroll
    for (int v = 0; v < 8; ++v) {
        int rl = v + (hi ? 8 : 0);
        float ls = __shfl(l_run, rl, 32);
        float invl = 1.0f / ls;
        size_t orow = ((size_t)b * NQ + n0 + rl) * BDIM + (size_t)h * HDIM;
#pragma unroll
        for (int t = 0; t < 8; ++t)
            aout[orow + t * 16 + l15] = (bf16)(o[t][v] * invl);
    }
}

// ---------------- host launcher ----------------
extern "C" void kernel_launch(void* const* d_in, const int* in_sizes, int n_in,
                              void* d_out, int out_size, void* d_ws, size_t ws_size,
                              hipStream_t stream) {
    const float* x    = (const float*)d_in[0];
    const float* ctx  = (const float*)d_in[1];
    const float* wq   = (const float*)d_in[2];
    const float* bq   = (const float*)d_in[3];
    const float* wkv  = (const float*)d_in[4];
    const float* bkv  = (const float*)d_in[5];
    const float* wo   = (const float*)d_in[6];
    const float* bo   = (const float*)d_in[7];
    const float* qns  = (const float*)d_in[8];
    float* out = (float*)d_out;

    char* p = (char*)d_ws;
    auto carve = [&](size_t bytes) -> bf16* {
        bf16* r = (bf16*)p;
        p += (bytes + 255) & ~(size_t)255;
        return r;
    };
    bf16* xb   = carve((size_t)RQ  * BDIM * 2);
    bf16* cb   = carve((size_t)RKV * BDIM * 2);
    bf16* wqt  = carve((size_t)BDIM * BDIM * 2);
    bf16* wkvt = carve((size_t)BDIM * 2 * BDIM * 2);
    bf16* wot  = carve((size_t)BDIM * BDIM * 2);
    bf16* qh   = carve((size_t)RQ  * BDIM * 2);      // [B,H,N,HD]
    bf16* kh   = carve((size_t)RKV * BDIM * 2);      // [B,H,M,HD]
    bf16* vth  = carve((size_t)RKV * BDIM * 2);      // [B,H,HD,M]
    bf16* ao   = carve((size_t)RQ  * BDIM * 2);      // [B*N, 1536]

    // casts
    {
        int n = RQ * BDIM;
        cast_bf16_kernel<<<(n / 4 + 255) / 256, 256, 0, stream>>>(x, xb, n);
        n = RKV * BDIM;
        cast_bf16_kernel<<<(n / 4 + 255) / 256, 256, 0, stream>>>(ctx, cb, n);
    }
    // weight transposes
    transpose_bf16_kernel<<<(BDIM * BDIM + 255) / 256, 256, 0, stream>>>(wq, wqt, BDIM, BDIM);
    transpose_bf16_kernel<<<(BDIM * 2 * BDIM + 255) / 256, 256, 0, stream>>>(wkv, wkvt, BDIM, 2 * BDIM);
    transpose_bf16_kernel<<<(BDIM * BDIM + 255) / 256, 256, 0, stream>>>(wo, wot, BDIM, BDIM);

    // Q = x@wq + bq, RMSNorm per head, -> bf16 [B,H,N,HD]
    gemm_wmma<0><<<dim3(RQ / 128, BDIM / 128), 256, 0, stream>>>(
        xb, wqt, bq, BDIM, BDIM, NQ, nullptr, qh, qns, nullptr, nullptr);

    // KV = ctx@wkv + bkv -> K [B,H,M,HD], Vt [B,H,HD,M]
    gemm_wmma<1><<<dim3(RKV / 128, (2 * BDIM) / 128), 256, 0, stream>>>(
        cb, wkvt, bkv, BDIM, 2 * BDIM, MK, nullptr, nullptr, nullptr, kh, vth);

    // attention -> bf16 [B*N, 1536]
    attn_kernel<<<dim3(NQ / 16, HEADS, NB), 32, 0, stream>>>(qh, kh, vth, ao);

    // out = ao@wo + bo -> f32
    gemm_wmma<2><<<dim3(RQ / 128, BDIM / 128), 256, 0, stream>>>(
        ao, wot, bo, BDIM, BDIM, NQ, out, nullptr, nullptr, nullptr, nullptr);
}
